// AllPairsSimilarity_76029511074093
// MI455X (gfx1250) — matrix operand order repaired and checked by reading, not compiled, not run on hardware
//
#include <hip/hip_runtime.h>
#include <hip/hip_bf16.h>

// ---------------------------------------------------------------------------
// AllPairsSimilarity on MI455X (gfx1250, wave32, WMMA)
//
//   q: (128, 640, 32, 32) f32   s: (5, 640, 32, 32) f32   out: (128,) f32
//
//   proto  = mean_shots(s)            -> (1024 patches, 640 ch)
//   s_norm = l2norm(proto)  (f16, row-major [m][c], K-contiguous)  [workspace]
//   rq     = 1/max(||q_patch||, eps)  (f32)                        [workspace]
//   sim    = q_raw @ s_norm^T  via v_wmma_f32_16x16x32_f16 (f32 acc)
//   out[nq]= mean_rows( rq * max_cols(sim) )   (fused epilogue, atomics)
//
// GEMM kernel: per wave, all 20 A-fragments (16x640) are register-resident;
// column tiles are processed TWO at a time (independent accumulator chains)
// with 1-deep B prefetch so L2 latency overlaps WMMA execution.
// ---------------------------------------------------------------------------

typedef __attribute__((ext_vector_type(16))) _Float16 v16h;
typedef __attribute__((ext_vector_type(8)))  _Float16 v8h;
typedef __attribute__((ext_vector_type(8)))  float    v8f;

#define NQ    128
#define CH    640
#define NPAT  1024          // 32*32 patches
#define SHOTS 5
#define KT    32            // WMMA K per step (f16)
#define NKT   (CH / KT)     // 20 k-steps
#define LDSA_STRIDE 656     // 640 + 16 halfs pad (16B-aligned rows, bank spread)
#define EPS   1e-12f

// ---------------------------------------------------------------- zero output
__global__ void zero_out_kernel(float* __restrict__ out) {
    int t = blockIdx.x * blockDim.x + threadIdx.x;
    if (t < NQ) out[t] = 0.0f;
}

// ------------------------------------------------- prototype: mean + l2norm
// thread <-> support patch n (coalesced over n); stores f16 s_norm[n][c]
__global__ void proto_kernel(const float* __restrict__ s,
                             _Float16* __restrict__ snorm) {
    int n = blockIdx.x * blockDim.x + threadIdx.x;   // 0..1023
    if (n >= NPAT) return;
    float sumsq = 0.0f;
    for (int c = 0; c < CH; ++c) {
        float v = 0.0f;
        #pragma unroll
        for (int sh = 0; sh < SHOTS; ++sh)
            v += s[(size_t)sh * CH * NPAT + (size_t)c * NPAT + n];
        v *= (1.0f / SHOTS);
        sumsq += v * v;
    }
    float r = 1.0f / fmaxf(sqrtf(sumsq), EPS);
    for (int c = 0; c < CH; ++c) {
        float v = 0.0f;
        #pragma unroll
        for (int sh = 0; sh < SHOTS; ++sh)
            v += s[(size_t)sh * CH * NPAT + (size_t)c * NPAT + n];
        v *= (1.0f / SHOTS);
        snorm[(size_t)n * CH + c] = (_Float16)(v * r);
    }
}

// --------------------------------------------- query reciprocal patch norms
// thread <-> (image, patch); consecutive threads -> consecutive n (coalesced)
__global__ void qnorm_kernel(const float* __restrict__ q,
                             float* __restrict__ rnorm) {
    int t = blockIdx.x * blockDim.x + threadIdx.x;   // 0..128*1024-1
    int nq = t >> 10;
    int n  = t & (NPAT - 1);
    if (nq >= NQ) return;
    const float* base = q + (size_t)nq * CH * NPAT + n;
    float sumsq = 0.0f;
    for (int c = 0; c < CH; ++c) {
        float v = base[(size_t)c * NPAT];
        sumsq += v * v;
    }
    rnorm[t] = 1.0f / fmaxf(sqrtf(sumsq), EPS);
}

// ------------------------------------------------------- fused WMMA GEMM
// grid = (64 row-tiles, 128 images), block = 128 threads = 4 waves.
// Block: 16 query patches (rows). Wave w: column tiles w, w+4, ..., w+60,
// processed as pairs (w+8i, w+8i+4), i = 0..7.
__global__ void __launch_bounds__(128, 1)
sim_wmma_kernel(const float*    __restrict__ q,
                const _Float16* __restrict__ snorm,
                const float*    __restrict__ rnorm,
                float*          __restrict__ out) {
    __shared__ _Float16 ldsA[16 * LDSA_STRIDE];  // 16 rows x 640 ch (f16, padded)
    __shared__ float    ldsMax[4][16];

    const int nq = blockIdx.y;
    const int n0 = blockIdx.x * 16;              // first query patch of tile
    const int t  = threadIdx.x;

    // ---- stage A tile: q[nq][c][n0..n0+15] -> f16 LDS (row = patch, col = c)
    {
        const float* qbase = q + (size_t)nq * CH * NPAT + n0;
        const int row  = t & 15;                 // 16 consecutive n -> coalesced
        const int cgrp = t >> 4;                 // 8 channels per pass
        for (int c = cgrp; c < CH; c += 8)
            ldsA[row * LDSA_STRIDE + c] = (_Float16)qbase[(size_t)c * NPAT + row];
    }
    __syncthreads();

    const int wave = t >> 5;
    const int lane = t & 31;
    const int mrow = lane & 15;                  // A: M index / B: N index
    const int hi   = lane >> 4;                  // lane-half selector

    // ---- hoist all A fragments for K = 0..640 into registers (20 x v16h)
    // A 16x32 layout (matches B): lanes 0-15 M=lane (K 0-15)
    //                            lanes 16-31 M=lane-16 (K 16-31)
    v16h afrag[NKT];
    #pragma unroll
    for (int kt = 0; kt < NKT; ++kt) {
        const _Float16* ap = &ldsA[mrow * LDSA_STRIDE + kt * KT + 16 * hi];
        v8h lo = *(const v8h*)(ap);              // ds_load_b128
        v8h hh = *(const v8h*)(ap + 8);          // ds_load_b128
        afrag[kt] = __builtin_shufflevector(lo, hh,
                     0, 1, 2, 3, 4, 5, 6, 7, 8, 9, 10, 11, 12, 13, 14, 15);
    }

    float max8[8];
    #pragma unroll
    for (int r = 0; r < 8; ++r) max8[r] = -3.402823466e+38f;

    // ---- sweep column tiles (support patches), two independent streams/wave
    // 16-bit B 32x16 layout used here (same K order as A): lane = column N (mod 16); 16 contiguous K
    // halfs starting at k0 + 16*hi.  snorm row m is K-contiguous => 32B loads.
    for (int i = 0; i < 8; ++i) {
        const int ct0 = wave + 8 * i;            // stream 0 column tile
        const int ct1 = ct0 + 4;                 // stream 1 column tile
        const _Float16* bb0 = snorm + (size_t)(ct0 * 16 + mrow) * CH + 16 * hi;
        const _Float16* bb1 = snorm + (size_t)(ct1 * 16 + mrow) * CH + 16 * hi;

        v8f acc0 = {};
        v8f acc1 = {};
        v16h b0 = *(const v16h*)(bb0);
        v16h b1 = *(const v16h*)(bb1);
        #pragma unroll
        for (int kt = 0; kt < NKT; ++kt) {
            v16h p0, p1;
            if (kt + 1 < NKT) {                  // prefetch next K-step (both)
                p0 = *(const v16h*)(bb0 + (kt + 1) * KT);
                p1 = *(const v16h*)(bb1 + (kt + 1) * KT);
            }
            acc0 = __builtin_amdgcn_wmma_f32_16x16x32_f16(
                       false, afrag[kt], false, b0, (short)0, acc0, false, false);
            acc1 = __builtin_amdgcn_wmma_f32_16x16x32_f16(
                       false, afrag[kt], false, b1, (short)0, acc1, false, false);
            b0 = p0;
            b1 = p1;
        }
        // C/D layout: VGPR r -> row (r | 8+r), lane -> column m*16 + (lane&15)
        #pragma unroll
        for (int r = 0; r < 8; ++r)
            max8[r] = fmaxf(max8[r], fmaxf(acc0[r], acc1[r]));
    }

    // ---- reduce max over the 16 columns held across each lane-half
    #pragma unroll
    for (int r = 0; r < 8; ++r) {
        float v = max8[r];
        v = fmaxf(v, __shfl_xor(v, 8, 16));
        v = fmaxf(v, __shfl_xor(v, 4, 16));
        v = fmaxf(v, __shfl_xor(v, 2, 16));
        v = fmaxf(v, __shfl_xor(v, 1, 16));
        max8[r] = v;
    }
    if (lane == 0) {
        #pragma unroll
        for (int r = 0; r < 8; ++r) ldsMax[wave][r] = max8[r];
    }
    if (lane == 16) {
        #pragma unroll
        for (int r = 0; r < 8; ++r) ldsMax[wave][8 + r] = max8[r];
    }
    __syncthreads();

    // ---- cross-wave max, scale by 1/||q||, accumulate mean into out[nq]
    if (t < 16) {
        float m = fmaxf(fmaxf(ldsMax[0][t], ldsMax[1][t]),
                        fmaxf(ldsMax[2][t], ldsMax[3][t]));
        float scaled = m * rnorm[nq * NPAT + n0 + t] * (1.0f / (float)NPAT);
        atomicAdd(&out[nq], scaled);
    }
}

// ---------------------------------------------------------------------------
extern "C" void kernel_launch(void* const* d_in, const int* in_sizes, int n_in,
                              void* d_out, int out_size, void* d_ws, size_t ws_size,
                              hipStream_t stream) {
    const float* q = (const float*)d_in[0];   // (128, 640, 32, 32)
    const float* s = (const float*)d_in[1];   // (5, 640, 32, 32)
    float* out = (float*)d_out;               // (128,)

    // workspace: [ s_norm f16 : 1024*640*2 B ][ rnorm f32 : 128*1024*4 B ]
    _Float16* snorm = (_Float16*)d_ws;
    float*    rnorm = (float*)((char*)d_ws + (size_t)NPAT * CH * sizeof(_Float16));

    zero_out_kernel<<<1, 128, 0, stream>>>(out);

    proto_kernel<<<NPAT / 256, 256, 0, stream>>>(s, snorm);

    qnorm_kernel<<<(NQ * NPAT) / 256, 256, 0, stream>>>(q, rnorm);

    dim3 grid(NPAT / 16, NQ);   // (64 row-tiles, 128 images)
    sim_wmma_kernel<<<grid, 128, 0, stream>>>(q, snorm, rnorm, out);
}
